// slot_hgat_21698174779819
// MI455X (gfx1250) — matrix operand order repaired and checked by reading, not compiled, not run on hardware
//
#include <hip/hip_runtime.h>
#include <hip/hip_bf16.h>

typedef __attribute__((ext_vector_type(8)))  float  v8f;
typedef __attribute__((ext_vector_type(16))) __bf16 v16bf;

#define BATCH  16
#define SEQ_N  512
#define DIM    512
#define NINT   32
#define GHD    64
#define NHEAD  8
#define DK     64
#define OUTC   128
#define NLAYER 2

#define TM 64
#define TN 64
#define TK 32
#define LDS_STRIDE 40   // elements; 80B rows: 16B-aligned, conflict-free b128 reads

// round-to-nearest-even f32 -> bf16 (raw ushort)
__device__ __forceinline__ unsigned short f2bf(float x)
{
    unsigned int u = __float_as_uint(x);
    u += 0x7FFFu + ((u >> 16) & 1u);
    return (unsigned short)(u >> 16);
}
__device__ __forceinline__ float bf2f(unsigned short h)
{
    return __uint_as_float((unsigned int)h << 16);
}

union FragU { uint4 q[2]; v16bf v; };

// ---------------------------------------------------------------------------
// Generic batched GEMM, fp32-accurate via bf16 hi/lo split on
// V_WMMA_F32_16X16X32_BF16:  A*B ~= aH*bH + aH*bL + aL*bH  (f32 accum)
// C[b] = act(A[b] (MxK) @ B[b] (KxN or NxK if transB) + bias) + resid[b]
// Block: 128 threads = 4 waves; block tile 64x64; wave tile 32x32.
// ---------------------------------------------------------------------------
__launch_bounds__(128)
__global__ void hg_gemm_wmma(const float* __restrict__ A, const float* __restrict__ Bw,
                             const float* __restrict__ bias, const float* __restrict__ resid,
                             float* __restrict__ C,
                             int M, int Nc, int K, int lda, int ldb, int ldc,
                             long long sA, long long sB, long long sC, long long sR,
                             int transB, int act)
{
    __shared__ __attribute__((aligned(16))) unsigned short AsH[TM][LDS_STRIDE];
    __shared__ __attribute__((aligned(16))) unsigned short AsL[TM][LDS_STRIDE];
    __shared__ __attribute__((aligned(16))) unsigned short BtH[TN][LDS_STRIDE]; // [n][k]
    __shared__ __attribute__((aligned(16))) unsigned short BtL[TN][LDS_STRIDE];

    const int bz = blockIdx.z;
    A  += (long long)bz * sA;
    Bw += (long long)bz * sB;
    C  += (long long)bz * sC;
    if (resid) resid += (long long)bz * sR;

    const int tileN = blockIdx.x * TN;
    const int tileM = blockIdx.y * TM;
    const int tid   = threadIdx.x;
    const int wave  = tid >> 5;
    const int lane  = tid & 31;
    const int wm    = (wave >> 1) * 32;   // wave row offset in block tile
    const int wn    = (wave & 1) * 32;    // wave col offset in block tile
    const int l15   = lane & 15;
    const int hi    = lane >> 4;

    v8f zacc = {};
    v8f acc[2][2];
    acc[0][0] = zacc; acc[0][1] = zacc; acc[1][0] = zacc; acc[1][1] = zacc;

    for (int k0 = 0; k0 < K; k0 += TK) {
        // prefetch next A tile (speculative; dropped on bad translation)
        if (k0 + TK < K) {
            int pr = tileM + (tid & 63);
            if (pr < M) __builtin_prefetch(A + (long long)pr * lda + k0 + TK, 0, 1);
        }
        // stage A tile (64 x 32) as bf16 hi/lo
        for (int i = tid; i < TM * TK; i += 128) {
            int r = i >> 5, c = i & 31;
            int gr = tileM + r, gc = k0 + c;
            float x = (gr < M && gc < K) ? A[(long long)gr * lda + gc] : 0.f;
            unsigned short h = f2bf(x);
            AsH[r][c] = h;
            AsL[r][c] = f2bf(x - bf2f(h));
        }
        // stage B tile transposed into Bt[n][k] as bf16 hi/lo
        if (!transB) {
            for (int i = tid; i < TK * TN; i += 128) {
                int kr = i >> 6, nc = i & 63;                 // coalesced over n
                int gk = k0 + kr, gn = tileN + nc;
                float x = (gk < K && gn < Nc) ? Bw[(long long)gk * ldb + gn] : 0.f;
                unsigned short h = f2bf(x);
                BtH[nc][kr] = h;
                BtL[nc][kr] = f2bf(x - bf2f(h));
            }
        } else {
            for (int i = tid; i < TK * TN; i += 128) {
                int nc = i >> 5, kr = i & 31;                 // coalesced over k
                int gk = k0 + kr, gn = tileN + nc;
                float x = (gk < K && gn < Nc) ? Bw[(long long)gn * ldb + gk] : 0.f;
                unsigned short h = f2bf(x);
                BtH[nc][kr] = h;
                BtL[nc][kr] = f2bf(x - bf2f(h));
            }
        }
        __syncthreads();

        // fragments (ISA 7.12.2, 16-bit A 16x32 / B 32x16 wave32 layouts)
        v16bf aH[2], aL[2], bH[2], bL[2];
        #pragma unroll
        for (int im = 0; im < 2; ++im) {
            int m = wm + im * 16 + l15;
            FragU fh, fl;
            fh.q[0] = *(const uint4*)&AsH[m][hi * 8];
            fh.q[1] = *(const uint4*)&AsH[m][16 + hi * 8];
            fl.q[0] = *(const uint4*)&AsL[m][hi * 8];
            fl.q[1] = *(const uint4*)&AsL[m][16 + hi * 8];
            aH[im] = fh.v; aL[im] = fl.v;
        }
        #pragma unroll
        for (int in = 0; in < 2; ++in) {
            int n = wn + in * 16 + l15;
            FragU fh, fl;
            fh.q[0] = *(const uint4*)&BtH[n][hi * 8];
            fh.q[1] = *(const uint4*)&BtH[n][16 + hi * 8];
            fl.q[0] = *(const uint4*)&BtL[n][hi * 8];
            fl.q[1] = *(const uint4*)&BtL[n][16 + hi * 8];
            bH[in] = fh.v; bL[in] = fl.v;
        }
        #pragma unroll
        for (int im = 0; im < 2; ++im)
            #pragma unroll
            for (int in = 0; in < 2; ++in) {
                acc[im][in] = __builtin_amdgcn_wmma_f32_16x16x32_bf16(
                    false, aL[im], false, bH[in], (short)0, acc[im][in], false, false);
                acc[im][in] = __builtin_amdgcn_wmma_f32_16x16x32_bf16(
                    false, aH[im], false, bL[in], (short)0, acc[im][in], false, false);
                acc[im][in] = __builtin_amdgcn_wmma_f32_16x16x32_bf16(
                    false, aH[im], false, bH[in], (short)0, acc[im][in], false, false);
            }
        __syncthreads();
    }

    // epilogue: f32 C/D layout -> lane L, VGPR r: row = (L<16 ? r : 8+r), col = L%16
    const int hi8 = hi * 8;
    #pragma unroll
    for (int im = 0; im < 2; ++im)
        #pragma unroll
        for (int in = 0; in < 2; ++in) {
            int gn = tileN + wn + in * 16 + l15;
            if (gn >= Nc) continue;
            float bv = bias ? bias[gn] : 0.f;
            #pragma unroll
            for (int r = 0; r < 8; ++r) {
                int gm = tileM + wm + im * 16 + hi8 + r;
                if (gm >= M) continue;
                float x = acc[im][in][r] + bv;
                if (act == 1)      x = x > 0.f ? x : __expf(x) - 1.f;  // ELU
                else if (act == 2) x = x > 0.f ? x : 0.2f * x;         // LeakyReLU
                if (resid) x += resid[(long long)gm * ldc + gn];
                C[(long long)gm * ldc + gn] = x;
            }
        }
}

// ---------------------------------------------------------------------------
// GAT softmax: P[row,j] = softmax_j( adj>0 ? leaky(s1[row] + s2[j]) : -9e15 )
// one wave32 per row, cols <= 512
// ---------------------------------------------------------------------------
__global__ void hg_gat_softmax(float* __restrict__ P, const float* __restrict__ s1,
                               const float* __restrict__ s2, const int* __restrict__ adj,
                               int Mrows, int cols, long long s1b, long long s2b)
{
    int row = blockIdx.x;
    int b = row / Mrows;
    int i = row - b * Mrows;
    int lane = threadIdx.x;
    float s1v = s1[(long long)b * s1b + i];
    const float* s2r = s2 + (long long)b * s2b;
    const int*   ar  = adj + (long long)row * cols;
    float*       pr  = P   + (long long)row * cols;

    float v[16];
    float m = -3.4e38f;
    int cnt = 0;
    for (int j = lane; j < cols; j += 32, ++cnt) {
        float e = s1v + s2r[j];
        e = e > 0.f ? e : 0.2f * e;
        e = (ar[j] > 0) ? e : -9e15f;
        v[cnt] = e;
        m = fmaxf(m, e);
    }
    for (int off = 16; off; off >>= 1) m = fmaxf(m, __shfl_xor(m, off, 32));
    float s = 0.f;
    cnt = 0;
    for (int j = lane; j < cols; j += 32, ++cnt) { float p = __expf(v[cnt] - m); v[cnt] = p; s += p; }
    for (int off = 16; off; off >>= 1) s += __shfl_xor(s, off, 32);
    float inv = 1.f / s;
    cnt = 0;
    for (int j = lane; j < cols; j += 32, ++cnt) pr[j] = v[cnt] * inv;
}

// ---------------------------------------------------------------------------
// MHA softmax (in place): S = softmax_j( mask==0 ? -1e9 : S*0.125 )
// ---------------------------------------------------------------------------
__global__ void hg_mha_softmax(float* __restrict__ S, const int* __restrict__ mask, int cols)
{
    int row = blockIdx.x;
    int lane = threadIdx.x;
    float*     sr = S    + (long long)row * cols;
    const int* mr = mask + (long long)row * cols;

    float v[16];
    float m = -3.4e38f;
    int cnt = 0;
    for (int j = lane; j < cols; j += 32, ++cnt) {
        float e = (mr[j] == 0) ? -1e9f : sr[j] * 0.125f;   // dk=64 -> 1/sqrt(dk)=0.125
        v[cnt] = e;
        m = fmaxf(m, e);
    }
    for (int off = 16; off; off >>= 1) m = fmaxf(m, __shfl_xor(m, off, 32));
    float s = 0.f;
    cnt = 0;
    for (int j = lane; j < cols; j += 32, ++cnt) { float p = __expf(v[cnt] - m); v[cnt] = p; s += p; }
    for (int off = 16; off; off >>= 1) s += __shfl_xor(s, off, 32);
    float inv = 1.f / s;
    cnt = 0;
    for (int j = lane; j < cols; j += 32, ++cnt) sr[j] = v[cnt] * inv;
}

// ---------------------------------------------------------------------------
// LayerNorm over DIM=512 columns, one wave32 per row
// ---------------------------------------------------------------------------
__global__ void hg_layernorm(const float* __restrict__ X, const float* __restrict__ g,
                             const float* __restrict__ bb, float* __restrict__ Y)
{
    int row = blockIdx.x, lane = threadIdx.x;
    const float* xr = X + (long long)row * DIM;
    float v[16];
    float s = 0.f, ss = 0.f;
    #pragma unroll
    for (int k = 0; k < 16; ++k) {
        float x = xr[lane + 32 * k];
        v[k] = x; s += x; ss += x * x;
    }
    for (int off = 16; off; off >>= 1) { s += __shfl_xor(s, off, 32); ss += __shfl_xor(ss, off, 32); }
    float mean = s * (1.f / DIM);
    float var  = ss * (1.f / DIM) - mean * mean;
    float rstd = rsqrtf(var + 1e-6f);
    float* yr = Y + (long long)row * DIM;
    #pragma unroll
    for (int k = 0; k < 16; ++k) {
        int c = lane + 32 * k;
        yr[c] = (v[k] - mean) * rstd * g[c] + bb[c];
    }
}

// ---------------------------------------------------------------------------
// Rank-1 GAT score dots
// ---------------------------------------------------------------------------
__global__ void hg_gat_head_scores(const float* __restrict__ hg, const float* __restrict__ ga,
                                   float* __restrict__ s1, float* __restrict__ s2)
{
    int idx = blockIdx.x * blockDim.x + threadIdx.x;       // b*H*N + h*N + n
    if (idx >= BATCH * NHEAD * SEQ_N) return;
    int n = idx % SEQ_N;
    int h = (idx / SEQ_N) % NHEAD;
    int b = idx / (SEQ_N * NHEAD);
    const float* rp = hg + ((long long)(b * SEQ_N + n)) * DIM + h * GHD;
    const float* a1 = ga + h * 2 * GHD;
    const float* a2 = a1 + GHD;
    float x1 = 0.f, x2 = 0.f;
    for (int d = 0; d < GHD; ++d) { float hv = rp[d]; x1 += hv * a1[d]; x2 += hv * a2[d]; }
    s1[idx] = x1; s2[idx] = x2;
}

__global__ void hg_gat_out_scores(const float* __restrict__ h2, const float* __restrict__ oa,
                                  float* __restrict__ s1, float* __restrict__ s2)
{
    int idx = blockIdx.x * blockDim.x + threadIdx.x;       // b*N + n
    if (idx >= BATCH * SEQ_N) return;
    const float* rp = h2 + (long long)idx * DIM;
    const float* a1 = oa;
    const float* a2 = oa + DIM;
    float x1 = 0.f, x2 = 0.f;
    for (int d = 0; d < DIM; ++d) { float hv = rp[d]; x1 += hv * a1[d]; x2 += hv * a2[d]; }
    s1[idx] = x1; s2[idx] = x2;
}

// ---------------------------------------------------------------------------
// Elementwise helpers
// ---------------------------------------------------------------------------
__global__ void hg_copy(float* __restrict__ d, const float* __restrict__ s, int n)
{ int i = blockIdx.x * blockDim.x + threadIdx.x; if (i < n) d[i] = s[i]; }

__global__ void hg_bcast(float* __restrict__ d, const float* __restrict__ s, int per, int n)
{ int i = blockIdx.x * blockDim.x + threadIdx.x; if (i < n) d[i] = s[i % per]; }

__global__ void hg_add(float* __restrict__ o, const float* __restrict__ a,
                       const float* __restrict__ b, int n)
{ int i = blockIdx.x * blockDim.x + threadIdx.x; if (i < n) o[i] = a[i] + b[i]; }

__global__ void hg_add_relu(float* __restrict__ o, const float* __restrict__ a,
                            const float* __restrict__ b, int n)
{ int i = blockIdx.x * blockDim.x + threadIdx.x; if (i < n) { float x = a[i] + b[i]; o[i] = x > 0.f ? x : 0.f; } }

// ---------------------------------------------------------------------------
// Host-side helpers
// ---------------------------------------------------------------------------
static inline void launch_gemm(hipStream_t stream,
    const float* A, const float* Bw, const float* bias, const float* resid, float* C,
    int M, int Nc, int K, int lda, int ldb, int ldc,
    long long sA, long long sB, long long sC, long long sR,
    int transB, int act)
{
    dim3 grid((Nc + TN - 1) / TN, (M + TM - 1) / TM, BATCH);
    hg_gemm_wmma<<<grid, dim3(128), 0, stream>>>(A, Bw, bias, resid, C,
        M, Nc, K, lda, ldb, ldc, sA, sB, sC, sR, transB, act);
}

static void run_mha(hipStream_t stream, const float* q, const float* kv, int Lq, int Lk,
                    const int* mask, const float* w, const float* g, const float* bvec,
                    float* out, float* qh, float* kh, float* vh, float* o, float* o2, float* S)
{
    const long long DD = (long long)DIM * DIM;
    const float *wq = w, *wk = w + DD, *wv = w + 2 * DD, *wf = w + 3 * DD;
    long long sq = (long long)Lq * DIM, sk = (long long)Lk * DIM, ss = (long long)Lq * Lk;

    launch_gemm(stream, q,  wq, nullptr, nullptr, qh, Lq, DIM, DIM, DIM, DIM, DIM, sq, 0, sq, 0, 0, 0);
    launch_gemm(stream, kv, wk, nullptr, nullptr, kh, Lk, DIM, DIM, DIM, DIM, DIM, sk, 0, sk, 0, 0, 0);
    launch_gemm(stream, kv, wv, nullptr, nullptr, vh, Lk, DIM, DIM, DIM, DIM, DIM, sk, 0, sk, 0, 0, 0);
    for (int h = 0; h < NHEAD; ++h) {
        // S = Q_h @ K_h^T
        launch_gemm(stream, qh + h * DK, kh + h * DK, nullptr, nullptr, S,
                    Lq, Lk, DK, DIM, DIM, Lk, sq, sk, ss, 0, 1, 0);
        hg_mha_softmax<<<BATCH * Lq, 32, 0, stream>>>(S, mask, Lk);
        // o_h = S @ V_h  (written into its 64-col slice)
        launch_gemm(stream, S, vh + h * DK, nullptr, nullptr, o + h * DK,
                    Lq, DK, Lk, Lk, DIM, DIM, ss, sk, sq, 0, 0, 0);
    }
    // o2 = o @ wf + q (residual), then post-LN
    launch_gemm(stream, o, wf, nullptr, q, o2, Lq, DIM, DIM, DIM, DIM, DIM, sq, 0, sq, sq, 0, 0);
    hg_layernorm<<<BATCH * Lq, 32, 0, stream>>>(o2, g, bvec, out);
}

extern "C" void kernel_launch(void* const* d_in, const int* in_sizes, int n_in,
                              void* d_out, int out_size, void* d_ws, size_t ws_size,
                              hipStream_t stream)
{
    (void)in_sizes; (void)n_in; (void)out_size; (void)ws_size;
    const float* enc    = (const float*)d_in[0];
    const int*   adj_i  = (const int*)d_in[2];
    const int*   adj_s  = (const int*)d_in[3];
    const int*   adj_is = (const int*)d_in[4];
    const int*   adj_si = (const int*)d_in[5];
    const float* intent = (const float*)d_in[6];
    const float* gatW   = (const float*)d_in[7];
    const float* gatA   = (const float*)d_in[8];
    const float* goutW  = (const float*)d_in[9];
    const float* goutA  = (const float*)d_in[10];
    const float* mhaW   = (const float*)d_in[11];
    const float* lnG    = (const float*)d_in[12];
    const float* lnB    = (const float*)d_in[13];
    const float* W1     = (const float*)d_in[14];
    const float* b1     = (const float*)d_in[15];
    const float* W2     = (const float*)d_in[16];
    const float* b2     = (const float*)d_in[17];

    const size_t BND  = (size_t)BATCH * SEQ_N * DIM;   // 4,194,304 floats
    const size_t BNID = (size_t)BATCH * NINT * DIM;    // 262,144
    const size_t BHN  = (size_t)BATCH * NHEAD * SEQ_N; // 65,536
    const size_t BN   = (size_t)BATCH * SEQ_N;         // 8,192
    const long long ND  = (long long)SEQ_N * DIM;
    const long long NN  = (long long)SEQ_N * SEQ_N;

    float* w = (float*)d_ws;
    float* slot[9];
    for (int s = 0; s < 9; ++s) slot[s] = w + (size_t)s * BND;
    float* s1g = w + 9 * BND;
    float* s2g = s1g + BHN;
    float* s1o = s2g + BHN;
    float* s2o = s1o + BN;
    float* haA = s2o + BN;        // h_a current
    float* haB = haA + BNID;      // a2a result
    float* haC = haB + BNID;      // b2a result

    float* out_slot   = (float*)d_out;
    float* out_logits = out_slot + BND;
    float* out_graph  = out_logits + (size_t)BATCH * SEQ_N * OUTC;

    // ============== Phase A: GAT (slot_graph_out) ==============
    float* h_gat = slot[0];
    float* xc    = slot[1];
    float* P     = slot[2];
    float* h2    = slot[3];

    // per-head projection h = x @ W_h, concatenated head-major into h_gat
    for (int h = 0; h < NHEAD; ++h)
        launch_gemm(stream, enc, gatW + (size_t)h * DIM * GHD, nullptr, nullptr, h_gat + h * GHD,
                    SEQ_N, GHD, DIM, DIM, GHD, DIM, ND, 0, ND, 0, 0, 0);
    hg_gat_head_scores<<<(BATCH * NHEAD * SEQ_N + 255) / 256, 256, 0, stream>>>(h_gat, gatA, s1g, s2g);

    for (int h = 0; h < NHEAD; ++h) {
        hg_gat_softmax<<<BATCH * SEQ_N, 32, 0, stream>>>(P, s1g + h * SEQ_N, s2g + h * SEQ_N,
                adj_s, SEQ_N, SEQ_N, (long long)NHEAD * SEQ_N, (long long)NHEAD * SEQ_N);
        // xc_h = elu(P @ h_gat_h)
        launch_gemm(stream, P, h_gat + h * GHD, nullptr, nullptr, xc + h * GHD,
                    SEQ_N, GHD, SEQ_N, SEQ_N, DIM, DIM, NN, ND, ND, 0, 0, 1);
    }
    // output GAT layer
    launch_gemm(stream, xc, goutW, nullptr, nullptr, h2, SEQ_N, DIM, DIM, DIM, DIM, DIM, ND, 0, ND, 0, 0, 0);
    hg_gat_out_scores<<<(BATCH * SEQ_N + 255) / 256, 256, 0, stream>>>(h2, goutA, s1o, s2o);
    hg_gat_softmax<<<BATCH * SEQ_N, 32, 0, stream>>>(P, s1o, s2o, adj_s, SEQ_N, SEQ_N,
            (long long)SEQ_N, (long long)SEQ_N);
    // slot_graph_out = elu(P @ h2) + x
    launch_gemm(stream, P, h2, nullptr, enc, out_graph, SEQ_N, DIM, SEQ_N, SEQ_N, DIM, DIM,
                NN, ND, ND, ND, 0, 1);

    // ============== Phase B: MHA stack ==============
    float* hB = slot[0];
    hg_copy<<<(int)((BND + 255) / 256), 256, 0, stream>>>(hB, enc, (int)BND);
    hg_bcast<<<(int)((BNID + 255) / 256), 256, 0, stream>>>(haA, intent, NINT * DIM, (int)BNID);

    float* qh = slot[1], * kh = slot[2], * vh = slot[3];
    float* ob = slot[4], * o2 = slot[5], * S = slot[6];
    float* b2b = slot[7], * a2b = slot[8];

    const long long DD4 = 4LL * DIM * DIM;
    for (int i = 0; i < NLAYER; ++i) {
        const float* wl = mhaW + (size_t)i * 4 * DD4;
        const float* gl = lnG + (size_t)i * 4 * DIM;
        const float* bl = lnB + (size_t)i * 4 * DIM;
        // a2a(h_a), b2b(h_b), a2b(q=h_b,k=h_a), b2a(q=h_a,k=h_b)
        run_mha(stream, haA, haA, NINT,  NINT,  adj_i,  wl + 0 * DD4, gl + 0 * DIM, bl + 0 * DIM, haB, qh, kh, vh, ob, o2, S);
        run_mha(stream, hB,  hB,  SEQ_N, SEQ_N, adj_s,  wl + 1 * DD4, gl + 1 * DIM, bl + 1 * DIM, b2b, qh, kh, vh, ob, o2, S);
        run_mha(stream, hB,  haA, SEQ_N, NINT,  adj_is, wl + 2 * DD4, gl + 2 * DIM, bl + 2 * DIM, a2b, qh, kh, vh, ob, o2, S);
        run_mha(stream, haA, hB,  NINT,  SEQ_N, adj_si, wl + 3 * DD4, gl + 3 * DIM, bl + 3 * DIM, haC, qh, kh, vh, ob, o2, S);
        hg_add_relu<<<(int)((BNID + 255) / 256), 256, 0, stream>>>(haA, haB, haC, (int)BNID);
        hg_add_relu<<<(int)((BND + 255) / 256), 256, 0, stream>>>(hB, b2b, a2b, (int)BND);
    }

    // ============== Phase C: outputs ==============
    // slot_out = encoded + h_b
    hg_add<<<(int)((BND + 255) / 256), 256, 0, stream>>>(out_slot, enc, hB, (int)BND);
    // z = leaky(slot_out @ W1 + b1)
    float* z = slot[1];
    launch_gemm(stream, out_slot, W1, b1, nullptr, z, SEQ_N, DIM, DIM, DIM, DIM, DIM, ND, 0, ND, 0, 0, 2);
    // logits = z @ W2 + b2
    launch_gemm(stream, z, W2, b2, nullptr, out_logits, SEQ_N, OUTC, DIM, DIM, OUTC, OUTC,
                ND, 0, (long long)SEQ_N * OUTC, 0, 0, 0);
}
